// Attention_76158360092997
// MI455X (gfx1250) — matrix operand order repaired and checked by reading, not compiled, not run on hardware
//
#include <hip/hip_runtime.h>

// ---------------------------------------------------------------------------
// GQA attention forward for MI455X (gfx1250, wave32, WMMA f16 -> f32 acc).
// B=2, T=2048, DIM=1024, H=16, KV=4, HD=64, causal, RoPE theta=1e4.
// ---------------------------------------------------------------------------

typedef __attribute__((ext_vector_type(16))) _Float16 v16h;
typedef __attribute__((ext_vector_type(8)))  _Float16 v8h;
typedef __attribute__((ext_vector_type(8)))  float    v8f;
typedef __attribute__((ext_vector_type(4)))  float    v4f;

#define B_   2
#define T_   2048
#define DIM_ 1024
#define H_   16
#define KV_  4
#define HD_  64
#define NROW (B_ * T_)          // 4096 GEMM rows
#define LN_THETA 9.210340371976184f

// ---------------- WMMA helpers -------------------------------------------

__device__ __forceinline__ v8f wmma16(v16h a, v16h b, v8f c) {
  // D = A(16x32 f16) x B(32x16 f16) + C(16x16 f32)
  return __builtin_amdgcn_wmma_f32_16x16x32_f16(
      /*neg_a=*/false, a, /*neg_b=*/false, b,
      /*c_mod=*/(short)0, c, /*reuse_a=*/false, /*reuse_b=*/false);
}

// A-matrix fragment (16x32, row major in memory, row stride rs elements).
// ISA layout: lane L: M = L&15, K = 8*(L>>4) + e + (e>=8 ? 8 : 0)
__device__ __forceinline__ v16h load_a_frag(const _Float16* __restrict__ base,
                                            int row0, int rs, int k0, int lane) {
  const int m  = lane & 15;
  const int kb = (lane >> 4) << 3;                 // 0 or 8
  const _Float16* p = base + (size_t)(row0 + m) * rs + (size_t)(k0 + kb);
  v8h lo = *(const v8h*)(p);                        // K = kb .. kb+7
  v8h hi = *(const v8h*)(p + 16);                   // K = kb+16 .. kb+23
  v16h a;
#pragma unroll
  for (int i = 0; i < 8; ++i) { a[i] = lo[i]; a[i + 8] = hi[i]; }
  return a;
}

// B-matrix fragment (32x16) loaded from an N-major ("transposed") array
// baseT[N][K], row stride rs. ISA layout: lane L: N = L&15, K = 8*(L>>4) + e + (e>=8 ? 8 : 0)
__device__ __forceinline__ v16h load_b_frag(const _Float16* __restrict__ baseT,
                                            int col0, int rs, int k0, int lane) {
  const int n  = lane & 15;
  const int kb = (lane >> 4) << 3;                 // 0 or 8
  const _Float16* p = baseT + (size_t)(col0 + n) * rs + (size_t)(k0 + kb);
  v8h lo = *(const v8h*)(p);
  v8h hi = *(const v8h*)(p + 16);
  v16h b;
#pragma unroll
  for (int i = 0; i < 8; ++i) { b[i] = lo[i]; b[i + 8] = hi[i]; }
  return b;
}

__device__ __forceinline__ float rmax16(float v) {
#pragma unroll
  for (int o = 1; o < 16; o <<= 1) v = fmaxf(v, __shfl_xor(v, o, 32));
  return v;
}
__device__ __forceinline__ float rsum16(float v) {
#pragma unroll
  for (int o = 1; o < 16; o <<= 1) v += __shfl_xor(v, o, 32);
  return v;
}

// ---------------- conversion kernels -------------------------------------

__global__ __launch_bounds__(256) void cvt_x_kernel(const float* __restrict__ in,
                                                    _Float16* __restrict__ out,
                                                    int n4) {
  int i = blockIdx.x * blockDim.x + threadIdx.x;
  int stride = gridDim.x * blockDim.x;
  for (; i < n4; i += stride) {
    v4f x = *(const v4f*)(in + 4 * (size_t)i);
    _Float16* o = out + 4 * (size_t)i;
    o[0] = (_Float16)x[0]; o[1] = (_Float16)x[1];
    o[2] = (_Float16)x[2]; o[3] = (_Float16)x[3];
  }
}

// out[N][K] = (f16) in[K][N], LDS-tiled (coalesced both sides). K,N % 32 == 0.
__global__ __launch_bounds__(256) void transpose_cvt_kernel(const float* __restrict__ in,
                                                            _Float16* __restrict__ out,
                                                            int K, int N) {
  __shared__ _Float16 tile[32][33];
  const int kb = blockIdx.y * 32, nb = blockIdx.x * 32;
  const int tx = threadIdx.x, ty = threadIdx.y;     // 32 x 8
#pragma unroll
  for (int j = 0; j < 32; j += 8)
    tile[ty + j][tx] = (_Float16)in[(size_t)(kb + ty + j) * N + (nb + tx)];
  __syncthreads();
#pragma unroll
  for (int j = 0; j < 32; j += 8)
    out[(size_t)(nb + ty + j) * K + (kb + tx)] = tile[tx][ty + j];
}

// ---------------- fused QKV GEMM + RoPE ----------------------------------
// 48 col blocks of 32: [0,32)=Q, [32,40)=K, [40,48)=V.
// One 32x32 output tile per wave (2x2 WMMA register blocking).

__device__ __forceinline__ void qkv_store_tile(
    v8f acc, int type, int rbase, int n, int lane,
    _Float16* __restrict__ qh, _Float16* __restrict__ kh,
    _Float16* __restrict__ vhT) {
  const int hf = lane >> 4;
#pragma unroll
  for (int r = 0; r < 8; ++r) {
    const int gr = rbase + r + 8 * hf;              // row in [0, B*T)
    const int bb = gr >> 11;                        // / T_
    const int t  = gr & (T_ - 1);
    const float v = acc[r];
    if (type == 2) {
      const int kv = n >> 6, d = n & 63;
      vhT[(((size_t)(bb * KV_ + kv) * HD_ + d) * T_) + t] = (_Float16)v;
    } else {
      const float p = __shfl_xor(v, 1, 32);         // RoPE partner column
      const int d = n & 63, i = d >> 1;
      const float freq = __expf(-(float)i * (2.0f * LN_THETA / (float)HD_));
      float sn, cs;
      sincosf((float)t * freq, &sn, &cs);
      const float o = (n & 1) ? (p * sn + v * cs) : (v * cs - p * sn);
      if (type == 0) {
        const int h = n >> 6;
        qh[(((size_t)(bb * H_ + h) * T_ + t) * HD_) + d] = (_Float16)(o * 0.125f); // fold 1/sqrt(64)
      } else {
        const int kv = n >> 6;
        kh[(((size_t)(bb * KV_ + kv) * T_ + t) * HD_) + d] = (_Float16)o;
      }
    }
  }
}

__global__ __launch_bounds__(256) void qkv_rope_kernel(
    const _Float16* __restrict__ xh,
    const _Float16* __restrict__ wqT, const _Float16* __restrict__ wkT,
    const _Float16* __restrict__ wvT,
    _Float16* __restrict__ qh, _Float16* __restrict__ kh,
    _Float16* __restrict__ vhT) {
  const int lane = threadIdx.x & 31;
  const int wave = threadIdx.x >> 5;
  const int ct   = blockIdx.x * 8 + wave;           // 0..47 (32-wide col blocks)
  const int row0 = blockIdx.y * 32;

  const _Float16* wT;
  int n0, type;
  if (ct < 32)      { type = 0; wT = wqT; n0 = ct * 32; }
  else if (ct < 40) { type = 1; wT = wkT; n0 = (ct - 32) * 32; }
  else              { type = 2; wT = wvT; n0 = (ct - 40) * 32; }

  v8f a00 = {}, a01 = {}, a10 = {}, a11 = {};
#pragma unroll 2
  for (int k0 = 0; k0 < DIM_; k0 += 32) {
    v16h fa0 = load_a_frag(xh, row0,      DIM_, k0, lane);
    v16h fa1 = load_a_frag(xh, row0 + 16, DIM_, k0, lane);
    v16h fb0 = load_b_frag(wT, n0,        DIM_, k0, lane);
    v16h fb1 = load_b_frag(wT, n0 + 16,   DIM_, k0, lane);
    a00 = wmma16(fa0, fb0, a00);
    a01 = wmma16(fa0, fb1, a01);
    a10 = wmma16(fa1, fb0, a10);
    a11 = wmma16(fa1, fb1, a11);
  }

  const int nl = lane & 15;
  qkv_store_tile(a00, type, row0,      n0 + nl,      lane, qh, kh, vhT);
  qkv_store_tile(a01, type, row0,      n0 + 16 + nl, lane, qh, kh, vhT);
  qkv_store_tile(a10, type, row0 + 16, n0 + nl,      lane, qh, kh, vhT);
  qkv_store_tile(a11, type, row0 + 16, n0 + 16 + nl, lane, qh, kh, vhT);
}

// ---------------- flash attention ----------------------------------------
// One wave per 16-row query tile; key blocks of 32; online softmax.
__global__ __launch_bounds__(256) void attn_kernel(
    const _Float16* __restrict__ qh, const _Float16* __restrict__ kh,
    const _Float16* __restrict__ vhT, _Float16* __restrict__ ctx) {
  __shared__ _Float16 sP[8][16][32];                // per-wave P staging (1KB)

  const int lane = threadIdx.x & 31;
  const int wave = threadIdx.x >> 5;
  const int flat = blockIdx.x * 8 + wave;           // 0..4095
  const int qt = flat & 127;                        // T_/16 tiles
  const int bh = flat >> 7;
  const int b = bh >> 4, h = bh & 15, kv = h >> 2;  // GQA: rep = 4
  const int t0 = qt * 16;
  const int hf = lane >> 4, nl = lane & 15;

  const _Float16* qbase = qh  + (size_t)(b * H_  + h)  * T_ * HD_;
  const _Float16* kbase = kh  + (size_t)(b * KV_ + kv) * T_ * HD_;
  const _Float16* vbase = vhT + (size_t)(b * KV_ + kv) * HD_ * T_;

  const v16h aq0 = load_a_frag(qbase, t0, HD_, 0, lane);
  const v16h aq1 = load_a_frag(qbase, t0, HD_, 32, lane);

  v8f o0 = {}, o1 = {}, o2 = {}, o3 = {};
  float m8[8], l8[8];
#pragma unroll
  for (int r = 0; r < 8; ++r) { m8[r] = -1e30f; l8[r] = 0.0f; }

  const v8f zero = {};
  const int smax = t0 + 15;
  for (int s0 = 0; s0 <= smax; s0 += 32) {
    // ---- scores: two 16x16 tiles over 32 keys ----
    v16h bk = load_b_frag(kbase, s0, HD_, 0, lane);
    v8f S0 = wmma16(aq0, bk, zero);
    bk = load_b_frag(kbase, s0, HD_, 32, lane);
    S0 = wmma16(aq1, bk, S0);
    bk = load_b_frag(kbase, s0 + 16, HD_, 0, lane);
    v8f S1 = wmma16(aq0, bk, zero);
    bk = load_b_frag(kbase, s0 + 16, HD_, 32, lane);
    S1 = wmma16(aq1, bk, S1);

    const bool need_mask = (s0 + 31) > t0;

    // ---- online softmax (per-row state; rows live in 16-lane halves) ----
#pragma unroll
    for (int r = 0; r < 8; ++r) {
      const int t = t0 + r + 8 * hf;
      float s0v = S0[r], s1v = S1[r];
      if (need_mask) {
        if (s0 + nl > t)      s0v = -1e30f;
        if (s0 + 16 + nl > t) s1v = -1e30f;
      }
      const float mx   = rmax16(fmaxf(s0v, s1v));
      const float mnew = fmaxf(m8[r], mx);
      const float corr = __expf(m8[r] - mnew);
      const float p0 = __expf(s0v - mnew);
      const float p1 = __expf(s1v - mnew);
      l8[r] = l8[r] * corr + rsum16(p0 + p1);
      m8[r] = mnew;
      o0[r] *= corr; o1[r] *= corr; o2[r] *= corr; o3[r] *= corr;
      const int m = r + 8 * hf;
      sP[wave][m][nl]      = (_Float16)p0;
      sP[wave][m][16 + nl] = (_Float16)p1;
    }

    // ---- reload P as A fragment (C-layout -> A-layout via LDS) ----
    {
      const int m = lane & 15, kb = (lane >> 4) << 3;
      const _Float16* sp = &sP[wave][0][0];
      v8h lo = *(const v8h*)(sp + m * 32 + kb);
      v8h hi = *(const v8h*)(sp + m * 32 + kb + 16);
      v16h pA;
#pragma unroll
      for (int i = 0; i < 8; ++i) { pA[i] = lo[i]; pA[i + 8] = hi[i]; }

      // ---- O += P @ V over the 64-wide head dim (4 tiles) ----
      v16h bv = load_b_frag(vbase, 0, T_, s0, lane);
      o0 = wmma16(pA, bv, o0);
      bv = load_b_frag(vbase, 16, T_, s0, lane);
      o1 = wmma16(pA, bv, o1);
      bv = load_b_frag(vbase, 32, T_, s0, lane);
      o2 = wmma16(pA, bv, o2);
      bv = load_b_frag(vbase, 48, T_, s0, lane);
      o3 = wmma16(pA, bv, o3);
    }
  }

  // ---- normalize and write context [B*T][H*HD] f16 ----
#pragma unroll
  for (int r = 0; r < 8; ++r) {
    const float inv = 1.0f / l8[r];
    const int t = t0 + r + 8 * hf;
    _Float16* row = ctx + (size_t)(b * T_ + t) * (H_ * HD_) + h * HD_;
    row[nl]      = (_Float16)(o0[r] * inv);
    row[16 + nl] = (_Float16)(o1[r] * inv);
    row[32 + nl] = (_Float16)(o2[r] * inv);
    row[48 + nl] = (_Float16)(o3[r] * inv);
  }
}

// ---------------- output projection --------------------------------------
// One 32x32 output tile per wave (2x2 WMMA register blocking), f32 out.
__global__ __launch_bounds__(256) void proj_kernel(const _Float16* __restrict__ ctx,
                                                   const _Float16* __restrict__ woT,
                                                   float* __restrict__ out) {
  const int lane = threadIdx.x & 31;
  const int wave = threadIdx.x >> 5;
  const int ct   = blockIdx.x * 8 + wave;           // 0..31 (32-wide col blocks)
  const int row0 = blockIdx.y * 32;
  const int n0   = ct * 32;

  v8f a00 = {}, a01 = {}, a10 = {}, a11 = {};
#pragma unroll 2
  for (int k0 = 0; k0 < DIM_; k0 += 32) {
    v16h fa0 = load_a_frag(ctx, row0,      DIM_, k0, lane);
    v16h fa1 = load_a_frag(ctx, row0 + 16, DIM_, k0, lane);
    v16h fb0 = load_b_frag(woT, n0,        DIM_, k0, lane);
    v16h fb1 = load_b_frag(woT, n0 + 16,   DIM_, k0, lane);
    a00 = wmma16(fa0, fb0, a00);
    a01 = wmma16(fa0, fb1, a01);
    a10 = wmma16(fa1, fb0, a10);
    a11 = wmma16(fa1, fb1, a11);
  }

  const int hf = lane >> 4, nl = lane & 15;
#pragma unroll
  for (int r = 0; r < 8; ++r) {
    float* row_lo = out + (size_t)(row0 + r + 8 * hf) * DIM_ + n0;
    float* row_hi = out + (size_t)(row0 + 16 + r + 8 * hf) * DIM_ + n0;
    row_lo[nl]      = a00[r];
    row_lo[16 + nl] = a01[r];
    row_hi[nl]      = a10[r];
    row_hi[16 + nl] = a11[r];
  }
}

// ---------------- launch --------------------------------------------------

extern "C" void kernel_launch(void* const* d_in, const int* in_sizes, int n_in,
                              void* d_out, int out_size, void* d_ws, size_t ws_size,
                              hipStream_t stream) {
  (void)in_sizes; (void)n_in; (void)out_size; (void)ws_size;
  const float* x  = (const float*)d_in[0];
  /* d_in[1] = mask: exactly tril, handled analytically */
  const float* wq = (const float*)d_in[2];
  const float* wk = (const float*)d_in[3];
  const float* wv = (const float*)d_in[4];
  const float* wo = (const float*)d_in[5];
  float* out = (float*)d_out;

  char* ws = (char*)d_ws;
  _Float16* xh  = (_Float16*)(ws);                      // 8 MB  [B*T][DIM]
  _Float16* wqT = (_Float16*)(ws + 8388608);            // 2 MB  [1024][1024]
  _Float16* wkT = (_Float16*)(ws + 10485760);           // .5 MB [256][1024]
  _Float16* wvT = (_Float16*)(ws + 11010048);           // .5 MB [256][1024]
  _Float16* woT = (_Float16*)(ws + 11534336);           // 2 MB  [1024][1024]
  _Float16* qh  = (_Float16*)(ws + 13631488);           // 8 MB  [B][H][T][HD]
  _Float16* kh  = (_Float16*)(ws + 22020096);           // 2 MB  [B][KV][T][HD]
  _Float16* vhT = (_Float16*)(ws + 24117248);           // 2 MB  [B][KV][HD][T]
  _Float16* ctx = (_Float16*)(ws + 26214400);           // 8 MB  [B*T][H*HD]

  // 1) precision conversion + weight transposes
  cvt_x_kernel<<<1024, 256, 0, stream>>>(x, xh, (B_ * T_ * DIM_) / 4);
  transpose_cvt_kernel<<<dim3(32, 32), dim3(32, 8), 0, stream>>>(wq, wqT, DIM_, H_ * HD_);
  transpose_cvt_kernel<<<dim3(8, 32),  dim3(32, 8), 0, stream>>>(wk, wkT, DIM_, KV_ * HD_);
  transpose_cvt_kernel<<<dim3(8, 32),  dim3(32, 8), 0, stream>>>(wv, wvT, DIM_, KV_ * HD_);
  transpose_cvt_kernel<<<dim3(32, 32), dim3(32, 8), 0, stream>>>(wo, woT, H_ * HD_, DIM_);

  // 2) fused QKV projection + RoPE (WMMA, 32x32 tiles/wave)
  qkv_rope_kernel<<<dim3(6, NROW / 32), 256, 0, stream>>>(xh, wqT, wkT, wvT, qh, kh, vhT);

  // 3) flash attention (WMMA + online softmax)
  attn_kernel<<<dim3((B_ * H_ * (T_ / 16)) / 8), 256, 0, stream>>>(qh, kh, vhT, ctx);

  // 4) output projection (WMMA, 32x32 tiles/wave, f32 out) — full 32 col tiles
  proj_kernel<<<dim3(4, NROW / 32), 256, 0, stream>>>(ctx, woT, out);
}
